// DetectNet_78778290143828
// MI455X (gfx1250) — compile-verified, matrix operand order checked by reading, not run on hardware
//
#include <hip/hip_runtime.h>
#include <hip/hip_bf16.h>

typedef __attribute__((ext_vector_type(16))) _Float16 v16h;
typedef __attribute__((ext_vector_type(8)))  float    v8f;
typedef __attribute__((ext_vector_type(2)))  float    v2f;

// ---- workspace layout (float offsets) ----
#define WS_A      0          // 9,324,800  (conv1 out; reused conv2/conv3 out)
#define WS_B      9324800    // 2,306,304  (pool1 out)
#define WS_C      11631104   // 1,079,808  (pool2 out)
#define WS_W16F   12710912   // 47,808 floats = 95,616 halves (f16 conv weights)
#define WS_D1ACC  12758720   // 4096
#define WS_D1OUT  12762816   // 4096
#define WS_D2OUT  12766912   // 307,200
#define WS_SIDX   13074112   // 115,200 ints
// total ~52.8 MB

__device__ __forceinline__ float sigm_(float x) { return 1.0f / (1.0f + expf(-x)); }

// K index held by half h of this lane for the 16-bit 16x32 A/B fragment (ISA 7.12.2)
__device__ __forceinline__ int kmap16_(int h, int hi) {
    int v = h >> 1;
    int base = (v < 4) ? (v * 2) : (16 + (v - 4) * 2);
    return base + (h & 1) + (hi ? 8 : 0);
}

// ---------------- prep: f32->f16 conv weights + zero dense1 accumulator ----------------
__global__ void prep_kernel(const float* __restrict__ w1, const float* __restrict__ w2,
                            const float* __restrict__ w3, _Float16* __restrict__ w16,
                            float* __restrict__ d1acc) {
    int i = blockIdx.x * blockDim.x + threadIdx.x;
    if (i < 3456)                w16[i] = (_Float16)w1[i];
    else if (i < 3456 + 18432)   w16[i] = (_Float16)w2[i - 3456];
    else if (i < 95616)          w16[i] = (_Float16)w3[i - 21888];
    else if (i < 95616 + 4096)   d1acc[i - 95616] = 0.0f;
}

// ---------------- implicit-GEMM conv + bias + ReLU, f16 WMMA 16x16x32 ----------------
// in:  NHWC f32   wh: [KH][KW][CI][CO] f16   out: NHWC f32
__global__ void conv_wmma_kernel(const float* __restrict__ in, const _Float16* __restrict__ wh,
                                 const float* __restrict__ bias, float* __restrict__ out,
                                 int Bn, int IH, int IW, int CI, int KH, int KW,
                                 int OH, int OW, int CO, int MT, int NT) {
    int wave = blockIdx.x * (blockDim.x >> 5) + (threadIdx.x >> 5);
    int lane = threadIdx.x & 31;
    int tilesPerImg = MT * NT;
    if (wave >= Bn * tilesPerImg) return;
    int b  = wave / tilesPerImg;
    int r  = wave - b * tilesPerImg;
    int mt = r / NT;
    int nt = r - mt * NT;
    int hi = lane >> 4;
    int ln = lane & 15;
    int KWCI = KW * CI;
    int K    = KH * KWCI;
    int K32  = (K + 31) & ~31;
    int M    = OH * OW;
    int m    = mt * 16 + ln;
    int oy   = m / OW;
    int ox   = m - oy * OW;
    bool mvalid = (m < M);
    const float* inb = in + (size_t)b * IH * IW * CI;
    int n = nt * 16 + ln;
    v8f acc = {};
    for (int k0 = 0; k0 < K32; k0 += 32) {
        v16h af, bf;
#pragma unroll
        for (int h = 0; h < 16; ++h) {
            int kk = k0 + kmap16_(h, hi);
            float av = 0.0f;
            if (mvalid && kk < K) {
                int kh  = kk / KWCI;
                int rem = kk - kh * KWCI;
                int kw  = rem / CI;
                int c   = rem - kw * CI;
                av = inb[((size_t)(oy + kh) * IW + (ox + kw)) * CI + c];
            }
            af[h] = (_Float16)av;
            bf[h] = (kk < K) ? wh[(size_t)kk * CO + n] : (_Float16)0.0f;
        }
        acc = __builtin_amdgcn_wmma_f32_16x16x32_f16(false, af, false, bf,
                                                     (short)0, acc, false, false);
    }
    float bv = bias[n];
#pragma unroll
    for (int v = 0; v < 8; ++v) {
        int mg = mt * 16 + v + (hi ? 8 : 0);
        if (mg < M) {
            float val = acc[v] + bv;
            out[((size_t)b * M + mg) * CO + n] = val > 0.0f ? val : 0.0f;
        }
    }
}

// ---------------- 2x2 max pool, stride 2, VALID ----------------
__global__ void pool_kernel(const float* __restrict__ in, float* __restrict__ out,
                            int Bn, int IH, int IW, int C, int OH, int OW) {
    size_t i = (size_t)blockIdx.x * blockDim.x + threadIdx.x;
    size_t total = (size_t)Bn * OH * OW * C;
    if (i >= total) return;
    int c = (int)(i % C);
    size_t r = i / C;
    int x = (int)(r % OW); r /= OW;
    int y = (int)(r % OH);
    int b = (int)(r / OH);
    const float* p0 = in + (((size_t)b * IH + 2 * y) * IW + 2 * x) * C + c;
    const float* p1 = p0 + (size_t)IW * C;
    out[i] = fmaxf(fmaxf(p0[0], p0[C]), fmaxf(p1[0], p1[C]));
}

// ---------------- dense1: split-K f32 WMMA 16x16x4, atomic accumulate ----------------
// X: 8(x16 padded) x K, Wt: K x N (N=256)
__global__ void dense1_wmma_kernel(const float* __restrict__ X, const float* __restrict__ Wt,
                                   float* __restrict__ accbuf, int K, int N,
                                   int quadsPerChunk, int nChunks) {
    int wave = blockIdx.x * (blockDim.x >> 5) + (threadIdx.x >> 5);
    int lane = threadIdx.x & 31;
    if (wave >= (N / 16) * nChunks) return;
    int nt = wave / nChunks;
    int ch = wave - nt * nChunks;
    int hi = lane >> 4;
    int ln = lane & 15;
    int n  = nt * 16 + ln;
    v8f acc = {};
    int kbase = ch * quadsPerChunk * 4;
    for (int q = 0; q < quadsPerChunk; ++q) {
        int k = kbase + q * 4 + (hi ? 2 : 0);
        v2f a, bb;
        a.x  = (ln < 8) ? X[(size_t)ln * K + k]     : 0.0f;
        a.y  = (ln < 8) ? X[(size_t)ln * K + k + 1] : 0.0f;
        bb.x = Wt[(size_t)k * N + n];
        bb.y = Wt[(size_t)(k + 1) * N + n];
        if ((q & 7) == 0)
            __builtin_prefetch(&Wt[(size_t)(k + 32) * N + n], 0, 1);
        acc = __builtin_amdgcn_wmma_f32_16x16x4_f32(false, a, false, bb,
                                                    (short)0, acc, false, false);
    }
#pragma unroll
    for (int v = 0; v < 8; ++v) {
        int ml = v + (hi ? 8 : 0);
        atomicAdd(&accbuf[ml * N + n], acc[v]);
    }
}

__global__ void d1_finish_kernel(const float* __restrict__ acc, const float* __restrict__ bias,
                                 float* __restrict__ out) {
    int i = blockIdx.x * blockDim.x + threadIdx.x;
    if (i < 16 * 256) {
        float v = acc[i] + bias[i & 255];
        out[i] = v > 0.0f ? v : 0.0f;
    }
}

// ---------------- dense2: f32 WMMA 16x16x4, K=256, N=38400, + bias ----------------
__global__ void dense2_wmma_kernel(const float* __restrict__ X, const float* __restrict__ Wt,
                                   const float* __restrict__ bias, float* __restrict__ out,
                                   int K, int N) {
    int wave = blockIdx.x * (blockDim.x >> 5) + (threadIdx.x >> 5);
    int lane = threadIdx.x & 31;
    if (wave >= N / 16) return;
    int nt = wave;
    int hi = lane >> 4;
    int ln = lane & 15;
    int n  = nt * 16 + ln;
    v8f acc = {};
    for (int q = 0; q < K / 4; ++q) {
        int k = q * 4 + (hi ? 2 : 0);
        v2f a, bb;
        a.x  = (ln < 8) ? X[ln * K + k]     : 0.0f;
        a.y  = (ln < 8) ? X[ln * K + k + 1] : 0.0f;
        bb.x = Wt[(size_t)k * N + n];
        bb.y = Wt[(size_t)(k + 1) * N + n];
        acc = __builtin_amdgcn_wmma_f32_16x16x4_f32(false, a, false, bb,
                                                    (short)0, acc, false, false);
    }
    float bv = bias[n];
#pragma unroll
    for (int v = 0; v < 8; ++v) {
        int ml = v + (hi ? 8 : 0);
        if (ml < 8) out[(size_t)ml * N + n] = acc[v] + bv;
    }
}

// ---------------- predict transform ----------------
__global__ void transform_kernel(const float* __restrict__ x, float* __restrict__ pred) {
    int cell = blockIdx.x * blockDim.x + threadIdx.x;   // (b, gy, gx) row-major, 8*40*60
    if (cell >= 8 * 40 * 60) return;
    int rr = cell % (40 * 60);
    int gy = rr / 60;
    int gx = rr - gy * 60;
    const float* p = x + (size_t)cell * 16;
    float* o = pred + (size_t)cell * 16;
    o[0]  = (sigm_(p[0]) + (float)gx) * 4.0f;
    o[1]  = (sigm_(p[1]) + (float)gy) * 4.0f;
    o[2]  = expf(p[2]) * 240.0f;   // 60 * 4
    o[3]  = expf(p[3]) * 120.0f;   // 30 * 4
    o[4]  = sigm_(p[4]);
    o[5]  = sigm_(p[5]);
    o[6]  = sigm_(p[6]);
    o[7]  = sigm_(p[7]);
    o[8]  = (p[8] + (float)gx) * 4.0f;
    o[9]  = (p[9] + (float)gy) * 4.0f;
    o[10] = expf(p[10]) * 80.0f;   // 20 * 4
    o[11] = expf(p[11]) * 160.0f;  // 40 * 4
    o[12] = p[12];
    o[13] = sigm_(p[13]);
    o[14] = sigm_(p[14]);
    o[15] = sigm_(p[15]);
}

// ---------------- NMS: stable descending rank-sort (matches argsort(-score)) ----------------
__global__ void nms_sort_kernel(const float* __restrict__ pred, int* __restrict__ sidx) {
    __shared__ float s[4800];
    int blk = blockIdx.x;                 // img*3 + cls
    int img = blk / 3;
    int cls = blk - img * 3;
    const float* pp = pred + (size_t)img * 38400;
    for (int n = threadIdx.x; n < 4800; n += blockDim.x) {
        int branch = n / 2400;
        int cellI  = n - branch * 2400;
        const float* f = pp + cellI * 16 + branch * 8;
        float conf = f[4];
        float c0 = f[5], c1 = f[6], c2 = f[7];
        int bi = 0; float best = c0;
        if (c1 > best) { best = c1; bi = 1; }
        if (c2 > best) { bi = 2; }
        s[n] = ((conf > 0.5f) && (bi == cls)) ? conf : -1.0f;
    }
    __syncthreads();
    int* so = sidx + (size_t)blk * 4800;
    for (int n = threadIdx.x; n < 4800; n += blockDim.x) {
        float sn = s[n];
        int rank = 0;
        for (int j = 0; j < 4800; ++j) {
            float sj = s[j];
            rank += ((sj > sn) || (sj == sn && j < n)) ? 1 : 0;
        }
        so[rank] = n;                     // sorted position -> original index
    }
}

__global__ void nms_main_kernel(const float* __restrict__ pred, const int* __restrict__ sidx,
                                float* __restrict__ keepOut) {
    const int NT = 256, PT = 19;
    __shared__ int bkeep;
    int blk = blockIdx.x;
    int img = blk / 3;
    int cls = blk - img * 3;
    const float* pp = pred + (size_t)img * 38400;
    const int* so = sidx + (size_t)blk * 4800;
    int t = threadIdx.x;
    float X1[PT], Y1[PT], X2[PT], Y2[PT], AR[PT];
    unsigned kb = 0;
#pragma unroll
    for (int p = 0; p < PT; ++p) {
        int sp = p * NT + t;
        X1[p] = 0.0f; Y1[p] = 0.0f; X2[p] = -1.0f; Y2[p] = -1.0f; AR[p] = 0.0f;
        if (sp < 4800) {
            int idx    = so[sp];
            int branch = idx / 2400;
            int cellI  = idx - branch * 2400;
            const float* f = pp + cellI * 16 + branch * 8;
            float bx = f[0], by = f[1], bw = f[2], bh = f[3], conf = f[4];
            float c0 = f[5], c1 = f[6], c2 = f[7];
            int bi = 0; float best = c0;
            if (c1 > best) { best = c1; bi = 1; }
            if (c2 > best) { bi = 2; }
            X1[p] = bx - bw * 0.5f; X2[p] = bx + bw * 0.5f;
            Y1[p] = by - bh * 0.5f; Y2[p] = by + bh * 0.5f;
            AR[p] = (X2[p] - X1[p] + 1.0f) * (Y2[p] - Y1[p] + 1.0f);
            if (conf > 0.5f && bi == cls) kb |= 1u << p;
        }
    }
    for (int i = 0; i < 4800; ++i) {
        __syncthreads();
        if (t == (i & (NT - 1))) bkeep = (int)((kb >> (i >> 8)) & 1u);
        __syncthreads();
        if (bkeep) {
            int idx    = so[i];
            int branch = idx / 2400;
            int cellI  = idx - branch * 2400;
            const float* f = pp + cellI * 16 + branch * 8;
            float bx = f[0], by = f[1], bw = f[2], bh = f[3];
            float bx1 = bx - bw * 0.5f, bx2 = bx + bw * 0.5f;
            float by1 = by - bh * 0.5f, by2 = by + bh * 0.5f;
            float ba  = (bx2 - bx1 + 1.0f) * (by2 - by1 + 1.0f);
#pragma unroll
            for (int p = 0; p < PT; ++p) {
                int sp = p * NT + t;
                if (sp > i && sp < 4800 && ((kb >> p) & 1u)) {
                    float iw = fminf(bx2, X2[p]) - fmaxf(bx1, X1[p]) + 1.0f;
                    float ih = fminf(by2, Y2[p]) - fmaxf(by1, Y1[p]) + 1.0f;
                    iw = iw > 0.0f ? iw : 0.0f;
                    ih = ih > 0.0f ? ih : 0.0f;
                    float inter = iw * ih;
                    float iou = inter / (ba + AR[p] - inter);
                    if (iou >= 0.4f) kb &= ~(1u << p);
                }
            }
        }
    }
#pragma unroll
    for (int p = 0; p < PT; ++p) {
        int sp = p * NT + t;
        if (sp < 4800) {
            int orig = so[sp];
            keepOut[(size_t)blk * 4800 + orig] = ((kb >> p) & 1u) ? 1.0f : 0.0f;
        }
    }
}

extern "C" void kernel_launch(void* const* d_in, const int* in_sizes, int n_in,
                              void* d_out, int out_size, void* d_ws, size_t ws_size,
                              hipStream_t stream) {
    (void)in_sizes; (void)n_in; (void)out_size; (void)ws_size;
    const float* img = (const float*)d_in[0];
    const float* w1  = (const float*)d_in[1];
    const float* b1  = (const float*)d_in[2];
    const float* w2  = (const float*)d_in[3];
    const float* b2  = (const float*)d_in[4];
    const float* w3  = (const float*)d_in[5];
    const float* b3  = (const float*)d_in[6];
    const float* wd1 = (const float*)d_in[7];
    const float* bd1 = (const float*)d_in[8];
    const float* wd2 = (const float*)d_in[9];
    const float* bd2 = (const float*)d_in[10];

    float* ws   = (float*)d_ws;
    float* bufA = ws + WS_A;
    float* bufB = ws + WS_B;
    float* bufC = ws + WS_C;
    _Float16* w16 = (_Float16*)(ws + WS_W16F);
    float* d1acc = ws + WS_D1ACC;
    float* d1out = ws + WS_D1OUT;
    float* d2out = ws + WS_D2OUT;
    int*   sidx  = (int*)(ws + WS_SIDX);
    float* pred  = (float*)d_out;               // 307,200 floats
    float* keep  = pred + 307200;               // 115,200 floats

    // weights -> f16, zero dense1 accumulator
    prep_kernel<<<(99712 + 255) / 256, 256, 0, stream>>>(w1, w2, w3, w16, d1acc);

    // conv1: 8x160x240x3 -> 8x155x235x32  (K=108, MT=2277, NT=2 -> 36432 waves)
    conv_wmma_kernel<<<4554, 256, 0, stream>>>(img, w16, b1, bufA,
                                               8, 160, 240, 3, 6, 6, 155, 235, 32, 2277, 2);
    pool_kernel<<<(2306304 + 255) / 256, 256, 0, stream>>>(bufA, bufB, 8, 155, 235, 32, 77, 117);

    // conv2: 8x77x117x32 -> 8x75x115x64  (K=288, MT=540, NT=4 -> 17280 waves)
    conv_wmma_kernel<<<2160, 256, 0, stream>>>(bufB, w16 + 3456, b2, bufA,
                                               8, 77, 117, 32, 3, 3, 75, 115, 64, 540, 4);
    pool_kernel<<<(1079808 + 255) / 256, 256, 0, stream>>>(bufA, bufC, 8, 75, 115, 64, 37, 57);

    // conv3: 8x37x57x64 -> 8x35x55x128  (K=576, MT=121, NT=8 -> 7744 waves)
    conv_wmma_kernel<<<968, 256, 0, stream>>>(bufC, w16 + 21888, b3, bufA,
                                              8, 37, 57, 64, 3, 3, 35, 55, 128, 121, 8);

    // dense1: (8x246400) @ (246400x256), split-K: 440 chunks x 140 quads, 16 N-tiles
    dense1_wmma_kernel<<<880, 256, 0, stream>>>(bufA, wd1, d1acc, 246400, 256, 140, 440);
    d1_finish_kernel<<<16, 256, 0, stream>>>(d1acc, bd1, d1out);

    // dense2: (8x256) @ (256x38400) + bias
    dense2_wmma_kernel<<<300, 256, 0, stream>>>(d1out, wd2, bd2, d2out, 256, 38400);

    // predict transform -> pred in d_out
    transform_kernel<<<75, 256, 0, stream>>>(d2out, pred);

    // NMS: 24 blocks = 8 images x 3 classes
    nms_sort_kernel<<<24, 256, 0, stream>>>(pred, sidx);
    nms_main_kernel<<<24, 256, 0, stream>>>(pred, sidx, keep);
}
